// ResTransformer_85237920956547
// MI455X (gfx1250) — matrix operand order repaired and checked by reading, not compiled, not run on hardware
//
#include <hip/hip_runtime.h>
#include <hip/hip_bf16.h>

#define N_NODES   20000
#define N_EDGES   100000
#define NH        256
#define MATSZ     (NH * NH)            // 65536 elems per weight matrix

typedef __bf16 bf16_t;
typedef __attribute__((ext_vector_type(16))) __bf16        v16bf;
typedef __attribute__((ext_vector_type(8)))  float         v8f;
typedef __attribute__((ext_vector_type(4)))  float         v4f;
typedef __attribute__((ext_vector_type(4)))  unsigned int  v4u;

union BF16Frag {
    v16bf bf;
    v4u   u4[2];
};

__device__ __forceinline__ float wave_sum(float v) {
    #pragma unroll
    for (int off = 16; off >= 1; off >>= 1) v += __shfl_xor(v, off, 32);
    return v;
}

// monotone order-preserving f32 -> u32 map (for atomic max on floats)
__device__ __forceinline__ unsigned fmap(float f) {
    unsigned u = __float_as_uint(f);
    return (u & 0x80000000u) ? ~u : (u | 0x80000000u);
}
__device__ __forceinline__ float funmap(unsigned u) {
    u = (u & 0x80000000u) ? (u & 0x7FFFFFFFu) : ~u;
    return __uint_as_float(u);
}

// LDS aperture maps addr[31:0] -> LDS byte offset (ISA 10.2), so truncation works.
__device__ __forceinline__ unsigned lds_off32(const void* p) {
    return (unsigned)(unsigned long long)(uintptr_t)p;
}

// ---------------- one-time weight prep ----------------
// All 16 layers x {Wq,Wk,Wv,Ws} -> bf16, transposed [N=256][K=256], K zero-padded
// (layer 0 has K=16). Lets the GEMM stage raw bytes with async-to-LDS copies.
__global__ __launch_bounds__(256) void k_wprep(
    const float* __restrict__ Wq1, const float* __restrict__ Wk1,
    const float* __restrict__ Wv1, const float* __restrict__ Ws1,
    const float* __restrict__ Wq,  const float* __restrict__ Wk,
    const float* __restrict__ Wv,  const float* __restrict__ Ws,
    bf16_t* __restrict__ wtb)
{
    size_t tid = (size_t)blockIdx.x * 256 + threadIdx.x;
    if (tid >= (size_t)64 * MATSZ) return;
    int mi = (int)(tid >> 16);         // matrix index 0..63
    int r  = (int)(tid & 65535);
    int n  = r >> 8, k = r & 255;
    int L = mi >> 2, m = mi & 3;
    const float* src; int Kd;
    if (L == 0) {
        src = (m == 0) ? Wq1 : (m == 1) ? Wk1 : (m == 2) ? Wv1 : Ws1;
        Kd = 16;
    } else {
        int l = L - 1;
        src = ((m == 0) ? Wq : (m == 1) ? Wk : (m == 2) ? Wv : Ws) + (size_t)l * MATSZ;
        Kd = NH;
    }
    float v = (k < Kd) ? src[(size_t)k * NH + n] : 0.f;
    wtb[tid] = (bf16_t)v;
}

// ---------------- layer-input prep ----------------

// Layer 0: hb[n, 0:16] = bf16(log(x+1)), hb[n, 16:256] = 0  (K-padded for uniform GEMM)
__global__ __launch_bounds__(256) void k_prep0(const float* __restrict__ x,
                                               bf16_t* __restrict__ hb) {
    int tid = blockIdx.x * 256 + threadIdx.x;
    if (tid >= N_NODES * NH) return;
    int n = tid >> 8, c = tid & 255;
    float v = 0.f;
    if (c < 16) v = logf(x[n * 16 + c] + 1.0f);
    hb[tid] = (bf16_t)v;
}

// Blocks 1..15: hb = bf16(relu(h / std(h, ddof=1)))   (one wave per node, float4 loads)
__global__ __launch_bounds__(256) void k_prep_block(const float* __restrict__ h,
                                                    bf16_t* __restrict__ hb) {
    int wave = threadIdx.x >> 5, lane = threadIdx.x & 31;
    int n = blockIdx.x * 8 + wave;
    if (n >= N_NODES) return;
    const float* row = h + (size_t)n * NH;
    int cb = lane * 8;
    v4f a = *(const v4f*)(row + cb);
    v4f b = *(const v4f*)(row + cb + 4);
    float sum = 0.f, sq = 0.f;
    #pragma unroll
    for (int j = 0; j < 4; ++j) {
        sum += a[j] + b[j];
        sq  += a[j] * a[j] + b[j] * b[j];
    }
    sum = wave_sum(sum);
    sq  = wave_sum(sq);
    float mean = sum * (1.0f / NH);
    float var  = (sq - NH * mean * mean) * (1.0f / (NH - 1));
    float inv  = rsqrtf(fmaxf(var, 1e-30f));
    union { bf16_t e[8]; v4u u; } pk;
    #pragma unroll
    for (int j = 0; j < 4; ++j) {
        pk.e[j]     = (bf16_t)fmaxf(a[j] * inv, 0.f);
        pk.e[j + 4] = (bf16_t)fmaxf(b[j] * inv, 0.f);
    }
    *(v4u*)(hb + (size_t)n * NH + cb) = pk.u;
}

// ---------------- WMMA GEMM: out[M,256] = hb[M,256] @ W + b ----------------
// grid.x = 157 row-blocks * 4 col-groups ; grid.y selects (Wq,Wk,Wv,Ws).
// Weights come pre-transposed bf16 [N][K]; the ENTIRE 64-col x 256-K panel
// (32 KB) is staged into LDS in one shot with global_load_async_to_lds_b128
// (ASYNCcnt), so the whole kernel needs exactly one wait + one barrier and the
// 32 WMMAs run back-to-back against clause'd ds_load_b128s.
__global__ __launch_bounds__(256) void k_gemm(
    const bf16_t* __restrict__ hb,
    const bf16_t* __restrict__ wtL,   // 4 matrices of [256][256] bf16 (transposed)
    const float* __restrict__ bq, const float* __restrict__ bk,
    const float* __restrict__ bv, const float* __restrict__ bs,
    float* __restrict__ qo, float* __restrict__ ko,
    float* __restrict__ vo, float* __restrict__ yo)
{
    const bf16_t* Wm = wtL + (size_t)blockIdx.y * MATSZ;
    const float* b; float* out;
    switch (blockIdx.y) {
        case 0:  b = bq; out = qo; break;
        case 1:  b = bk; out = ko; break;
        case 2:  b = bv; out = vo; break;
        default: b = bs; out = yo; break;
    }
    const int colg   = blockIdx.x & 3;     // 4 col-groups of 64
    const int rowblk = blockIdx.x >> 2;    // 128-row blocks
    const int wave = threadIdx.x >> 5, lane = threadIdx.x & 31;
    const int rowbase = rowblk * 128 + wave * 16;

    const int ROWB = 528;                  // padded LDS row stride (256*2 + 16 B)
    __shared__ __align__(16) char Wt[64 * ROWB];   // 33792 B
    const unsigned ldsbase = lds_off32(&Wt[0]);

    // async-stage the full weight panel: 64 rows x 512 B = 8 x b128 per thread
    {
        const int srow = threadIdx.x >> 2;                 // 0..63 (col within group)
        const int scol = threadIdx.x & 3;                  // 4 threads per row
        const bf16_t* grow = Wm + (((size_t)(colg * 64 + srow)) << 8) + scol * 64;
        const unsigned lrow = ldsbase + srow * ROWB + scol * 128;
        #pragma unroll
        for (int i = 0; i < 8; ++i) {
            asm volatile("global_load_async_to_lds_b128 %0, %1, off"
                         :: "v"(lrow + i * 16),
                            "v"((unsigned long long)(uintptr_t)(grow + i * 8))
                         : "memory");
        }
    }

    v8f acc[4] = {};

    // A-fragment addressing per ISA 16-bit 16x32 layout
    int m = rowbase + (lane & 15);
    if (m >= N_NODES) m = N_NODES - 1;
    const bf16_t* arow = hb + (size_t)m * NH;
    const int khalf = (lane >> 4) << 3;    // 0 or 8

    asm volatile("s_wait_asynccnt 0" ::: "memory");
    __syncthreads();                       // panel visible to all waves

    #pragma unroll
    for (int ks = 0; ks < 8; ++ks) {
        const int k0 = ks * 32;
        BF16Frag af;
        af.u4[0] = *(const v4u*)(arow + k0 + khalf);       // K = k0+0..7  / +8..15
        af.u4[1] = *(const v4u*)(arow + k0 + khalf + 16);  // K = k0+16..23 / +24..31
        const int klocB = (k0 + ((lane >> 4) << 3)) * 2;   // byte offset in LDS row
        #pragma unroll
        for (int ct = 0; ct < 4; ++ct) {
            const char* bp = Wt + (ct * 16 + (lane & 15)) * ROWB + klocB;
            BF16Frag bfr;
            bfr.u4[0] = *(const v4u*)(bp);
            bfr.u4[1] = *(const v4u*)(bp + 32);
            acc[ct] = __builtin_amdgcn_wmma_f32_16x16x32_bf16(
                false, af.bf, false, bfr.bf, (short)0, acc[ct], false, false);
        }
    }

    // store C + bias (VGPR i: lanes 0-15 -> M=i, lanes 16-31 -> M=8+i)
    const int r0 = rowbase + ((lane >> 4) << 3);
    #pragma unroll
    for (int ct = 0; ct < 4; ++ct) {
        int col = colg * 64 + ct * 16 + (lane & 15);
        float bias = b[col];
        #pragma unroll
        for (int i = 0; i < 8; ++i) {
            int row = r0 + i;
            if (row < N_NODES) out[(size_t)row * NH + col] = acc[ct][i] + bias;
        }
    }
}

// ---------------- edge phase ----------------

// alpha[e] = dot(q[dst], k[src] + e) / 16, e recomputed on the fly (one wave per edge)
__global__ __launch_bounds__(256) void k_edge_alpha(
    const int* __restrict__ ei, const float* __restrict__ ea,
    const float* __restrict__ We,
    const float* __restrict__ q, const float* __restrict__ kmat,
    float* __restrict__ alpha)
{
    int wave = threadIdx.x >> 5, lane = threadIdx.x & 31;
    int e = blockIdx.x * 8 + wave;
    if (e >= N_EDGES) return;
    int src = ei[e], dst = ei[N_EDGES + e];
    float l0 = logf(ea[e * 4 + 0] + 1.f), l1 = logf(ea[e * 4 + 1] + 1.f);
    float l2 = logf(ea[e * 4 + 2] + 1.f), l3 = logf(ea[e * 4 + 3] + 1.f);
    const float* qr = q    + (size_t)dst * NH;
    const float* kr = kmat + (size_t)src * NH;
    int cb = lane * 8;
    float sum = 0.f;
    #pragma unroll
    for (int hh = 0; hh < 2; ++hh) {
        int c = cb + hh * 4;
        v4f qq = *(const v4f*)(qr + c);
        v4f kk = *(const v4f*)(kr + c);
        v4f ee = l0 * (*(const v4f*)(We + c))
               + l1 * (*(const v4f*)(We + NH + c))
               + l2 * (*(const v4f*)(We + 2 * NH + c))
               + l3 * (*(const v4f*)(We + 3 * NH + c));
        #pragma unroll
        for (int j = 0; j < 4; ++j) sum += qq[j] * (kk[j] + ee[j]);
    }
    sum = wave_sum(sum);
    if (lane == 0) alpha[e] = sum * 0.0625f;  // 1/sqrt(256)
}

__global__ __launch_bounds__(256) void k_init_ms(unsigned* __restrict__ mbits,
                                                 float* __restrict__ s) {
    int n = blockIdx.x * 256 + threadIdx.x;
    if (n < N_NODES) { mbits[n] = 0u; s[n] = 0.f; }
}

__global__ __launch_bounds__(256) void k_edge_max(const int* __restrict__ ei,
                                                  const float* __restrict__ alpha,
                                                  unsigned* __restrict__ mbits) {
    int e = blockIdx.x * 256 + threadIdx.x;
    if (e >= N_EDGES) return;
    atomicMax(&mbits[ei[N_EDGES + e]], fmap(alpha[e]));
}

__global__ __launch_bounds__(256) void k_edge_exp(const int* __restrict__ ei,
                                                  float* __restrict__ alpha,
                                                  const unsigned* __restrict__ mbits,
                                                  float* __restrict__ s) {
    int e = blockIdx.x * 256 + threadIdx.x;
    if (e >= N_EDGES) return;
    int dst = ei[N_EDGES + e];
    float a = expf(alpha[e] - funmap(mbits[dst]));
    alpha[e] = a;
    atomicAdd(&s[dst], a);
}

// y[dst] += attn * (v[src] + e)   (one wave per edge, f32 atomics into L2-resident y)
__global__ __launch_bounds__(256) void k_edge_scatter(
    const int* __restrict__ ei, const float* __restrict__ ea,
    const float* __restrict__ We,
    const float* __restrict__ vmat, const float* __restrict__ alpha,
    const float* __restrict__ s, float* __restrict__ y)
{
    int wave = threadIdx.x >> 5, lane = threadIdx.x & 31;
    int e = blockIdx.x * 8 + wave;
    if (e >= N_EDGES) return;
    int src = ei[e], dst = ei[N_EDGES + e];
    float attn = alpha[e] / (s[dst] + 1e-16f);
    float l0 = logf(ea[e * 4 + 0] + 1.f), l1 = logf(ea[e * 4 + 1] + 1.f);
    float l2 = logf(ea[e * 4 + 2] + 1.f), l3 = logf(ea[e * 4 + 3] + 1.f);
    const float* vr = vmat + (size_t)src * NH;
    float* yr = y + (size_t)dst * NH;
    int cb = lane * 8;
    #pragma unroll
    for (int hh = 0; hh < 2; ++hh) {
        int c = cb + hh * 4;
        v4f vv = *(const v4f*)(vr + c);
        v4f ee = l0 * (*(const v4f*)(We + c))
               + l1 * (*(const v4f*)(We + NH + c))
               + l2 * (*(const v4f*)(We + 2 * NH + c))
               + l3 * (*(const v4f*)(We + 3 * NH + c));
        #pragma unroll
        for (int j = 0; j < 4; ++j)
            atomicAdd(&yr[c + j], attn * (vv[j] + ee[j]));
    }
}

// h = beta*h + y   (beta=0 for layer 0 assign, 1 for residual); float4 per thread
__global__ __launch_bounds__(256) void k_residual(float* __restrict__ h,
                                                  const float* __restrict__ y,
                                                  float beta) {
    int i = (blockIdx.x * 256 + threadIdx.x) * 4;
    if (i < N_NODES * NH) {
        v4f hv = *(const v4f*)(h + i);
        v4f yv = *(const v4f*)(y + i);
        hv = beta * hv + yv;
        *(v4f*)(h + i) = hv;
    }
}

// out[n] = sum_c (h[c]/sqrt(15)) * (sum_o Wl[c,o]*scale[o]) + sum_o bl[o]*scale[o]
__global__ __launch_bounds__(256) void k_head(const float* __restrict__ h,
                                              const float* __restrict__ Wl,
                                              const float* __restrict__ bl,
                                              const float* __restrict__ scale,
                                              float* __restrict__ out) {
    int wave = threadIdx.x >> 5, lane = threadIdx.x & 31;
    int n = blockIdx.x * 8 + wave;
    if (n >= N_NODES) return;
    const float inv = 0.25819888974716112567f;  // 1/sqrt(15)
    const float* row = h + (size_t)n * NH;
    int cb = lane * 8;
    float sum = 0.f;
    #pragma unroll
    for (int i = 0; i < 8; ++i) {
        int c = cb + i;
        float wc = 0.f;
        #pragma unroll
        for (int o = 0; o < 8; ++o) wc += Wl[c * 8 + o] * scale[o];
        sum += row[c] * inv * wc;
    }
    sum = wave_sum(sum);
    if (lane == 0) {
        float bsum = 0.f;
        #pragma unroll
        for (int o = 0; o < 8; ++o) bsum += bl[o] * scale[o];
        out[n] = sum + bsum;
    }
}

extern "C" void kernel_launch(void* const* d_in, const int* in_sizes, int n_in,
                              void* d_out, int out_size, void* d_ws, size_t ws_size,
                              hipStream_t stream)
{
    (void)in_sizes; (void)n_in; (void)out_size; (void)ws_size;
    const float* x   = (const float*)d_in[0];
    const int*   ei  = (const int*)d_in[1];
    const float* ea  = (const float*)d_in[2];
    const float* Wq1 = (const float*)d_in[3];  const float* bq1 = (const float*)d_in[4];
    const float* Wk1 = (const float*)d_in[5];  const float* bk1 = (const float*)d_in[6];
    const float* Wv1 = (const float*)d_in[7];  const float* bv1 = (const float*)d_in[8];
    const float* We1 = (const float*)d_in[9];
    const float* Ws1 = (const float*)d_in[10]; const float* bs1 = (const float*)d_in[11];
    const float* Wq  = (const float*)d_in[12]; const float* bq  = (const float*)d_in[13];
    const float* Wk  = (const float*)d_in[14]; const float* bk  = (const float*)d_in[15];
    const float* Wv  = (const float*)d_in[16]; const float* bv  = (const float*)d_in[17];
    const float* We  = (const float*)d_in[18];
    const float* Ws  = (const float*)d_in[19]; const float* bs  = (const float*)d_in[20];
    const float* Wl  = (const float*)d_in[21]; const float* bl  = (const float*)d_in[22];
    const float* scl = (const float*)d_in[23];

    const size_t NM = (size_t)N_NODES * NH;
    float*    h      = (float*)d_ws;
    float*    y      = h + NM;
    float*    qb     = y + NM;
    float*    kb     = qb + NM;
    float*    vb     = kb + NM;
    bf16_t*   hb     = (bf16_t*)(vb + NM);
    float*    alpha  = (float*)(hb + NM);
    unsigned* mbits  = (unsigned*)(alpha + N_EDGES);
    float*    sbuf   = (float*)(mbits + N_NODES);
    bf16_t*   wtb    = (bf16_t*)(sbuf + N_NODES);   // 64 x 256x256 bf16 (8.4 MB)

    const dim3 gGemm(157 * 4, 4);
    const int nodeWaveBlocks = (N_NODES + 7) / 8;
    const int edgeWaveBlocks = (N_EDGES + 7) / 8;
    const int edgeThrBlocks  = (N_EDGES + 255) / 256;
    const int nodeThrBlocks  = (N_NODES + 255) / 256;
    const int elemBlocks     = (int)((NM + 255) / 256);
    const int vecBlocks      = (int)((NM / 4 + 255) / 256);
    const int wprepBlocks    = (int)(((size_t)64 * MATSZ + 255) / 256);

    k_wprep<<<wprepBlocks, 256, 0, stream>>>(Wq1, Wk1, Wv1, Ws1, Wq, Wk, Wv, Ws, wtb);

    for (int L = 0; L < 16; ++L) {
        const float *pbq, *pbk, *pbv, *pbs, *pWe;
        float beta;
        if (L == 0) {
            k_prep0<<<elemBlocks, 256, 0, stream>>>(x, hb);
            pbq = bq1; pbk = bk1; pbv = bv1; pbs = bs1; pWe = We1;
            beta = 0.f;
        } else {
            int l = L - 1;
            k_prep_block<<<nodeWaveBlocks, 256, 0, stream>>>(h, hb);
            pbq = bq + (size_t)l * NH; pbk = bk + (size_t)l * NH;
            pbv = bv + (size_t)l * NH; pbs = bs + (size_t)l * NH;
            pWe = We + (size_t)l * 4 * NH;
            beta = 1.f;
        }
        k_gemm<<<gGemm, 256, 0, stream>>>(hb, wtb + (size_t)L * 4 * MATSZ,
                                          pbq, pbk, pbv, pbs,
                                          qb, kb, vb, y);
        k_edge_alpha<<<edgeWaveBlocks, 256, 0, stream>>>(ei, ea, pWe, qb, kb, alpha);
        k_init_ms<<<nodeThrBlocks, 256, 0, stream>>>(mbits, sbuf);
        k_edge_max<<<edgeThrBlocks, 256, 0, stream>>>(ei, alpha, mbits);
        k_edge_exp<<<edgeThrBlocks, 256, 0, stream>>>(ei, alpha, mbits, sbuf);
        k_edge_scatter<<<edgeWaveBlocks, 256, 0, stream>>>(ei, ea, pWe, vb, alpha, sbuf, y);
        k_residual<<<vecBlocks, 256, 0, stream>>>(h, y, beta);
    }
    k_head<<<nodeWaveBlocks, 256, 0, stream>>>(h, Wl, bl, scl, (float*)d_out);
}
